// GaussianAwarePatchCore_24464133718497
// MI455X (gfx1250) — matrix operand order repaired and checked by reading, not compiled, not run on hardware
//
#include <hip/hip_runtime.h>
#include <math.h>

typedef __attribute__((ext_vector_type(16))) __bf16 v16bf;
typedef __attribute__((ext_vector_type(8)))  __bf16 v8bf;
typedef __attribute__((ext_vector_type(8)))  float  v8f;

#define HIN   256
#define CR    256
#define CG    5
#define CF    261   // CR + CG
#define NPIX  2048  // B(2) * 32 * 32
#define NBANK 65536
#define KD    256

#define MBLK     32                    // query rows per block (2 WMMA M-tiles)
#define NCHUNKS  8                     // N split for occupancy
#define NMBLK    (NPIX / MBLK)         // 64
#define TILES_PER_CHUNK (NBANK / 16 / NCHUNKS)   // 512
#define TILES_PER_WAVE  (TILES_PER_CHUNK / 8)    // 64

__device__ __forceinline__ unsigned short bf16_rne(float f) {
    unsigned int u = __float_as_uint(f);
    unsigned int r = 0x7FFFu + ((u >> 16) & 1u);
    return (unsigned short)((u + r) >> 16);
}

// ---------------------------------------------------------------------------
// init candidate buffer (packed {d2bits, index}) to +inf
// ---------------------------------------------------------------------------
__global__ __launch_bounds__(256)
void init_kernel(unsigned long long* __restrict__ cand) {
    int i = blockIdx.x * 256 + threadIdx.x;
    if (i < NPIX) cand[i] = 0xFFFFFFFFFFFFFFFFull;
}

// ---------------------------------------------------------------------------
// bank prep: bf16-hi copy + f32 row norms. 65536 blocks x 256.
// ---------------------------------------------------------------------------
__global__ __launch_bounds__(256)
void bank_prep_kernel(const float* __restrict__ bank,
                      unsigned short* __restrict__ bhi,
                      float* __restrict__ bnorm) {
    __shared__ float red[256];
    int n = blockIdx.x, t = threadIdx.x;
    float v = bank[(size_t)n * KD + t];
    bhi[(size_t)n * KD + t] = bf16_rne(v);
    red[t] = v * v;
    __syncthreads();
    for (int s = 128; s > 0; s >>= 1) {
        if (t < s) red[t] += red[t + s];
        __syncthreads();
    }
    if (t == 0) bnorm[n] = red[0];
}

// ---------------------------------------------------------------------------
// features: conv(8x8 s8)+relu, antialiased bilinear geo, fusion GEMV.
// Emits f32 queries (for exact pass 2), bf16-hi queries (pass 1), weights.
// ---------------------------------------------------------------------------
__global__ __launch_bounds__(256)
void feature_kernel(const float* __restrict__ image,
                    const float* __restrict__ geo,
                    const float* __restrict__ bw,
                    const float* __restrict__ bb,
                    const float* __restrict__ fw,
                    const float* __restrict__ fb,
                    float* __restrict__ xq,
                    unsigned short* __restrict__ ahi,
                    float* __restrict__ wgt) {
    __shared__ float patch[192];
    __shared__ float u[CF];
    int p = blockIdx.x, t = threadIdx.x;
    int b = p >> 10;
    int y = (p >> 5) & 31;
    int x = p & 31;

    if (t < 192) {
        int ci = t >> 6, r = (t >> 3) & 7, cc = t & 7;
        patch[t] = image[((size_t)(b * 3 + ci) * HIN + (y * 8 + r)) * HIN + (x * 8 + cc)];
    } else if (t < 192 + CG) {
        // jax.image.resize bilinear (antialias): triangle kernel width 8,
        // per-dimension border renormalization.
        int c = t - 192;
        float cy = 8.f * y + 3.5f, cx = 8.f * x + 3.5f;
        float swx = 0.f;
        for (int kx = 8 * x - 4; kx <= 8 * x + 11; ++kx)
            if (kx >= 0 && kx < HIN) swx += 1.f - fabsf((float)kx - cx) * 0.125f;
        float acc = 0.f, swy = 0.f;
        for (int ky = 8 * y - 4; ky <= 8 * y + 11; ++ky) {
            if (ky < 0 || ky >= HIN) continue;
            float wy = 1.f - fabsf((float)ky - cy) * 0.125f;
            swy += wy;
            const float* row = geo + ((size_t)(b * CG + c) * HIN + ky) * HIN;
            for (int kx = 8 * x - 4; kx <= 8 * x + 11; ++kx) {
                if (kx < 0 || kx >= HIN) continue;
                acc += wy * (1.f - fabsf((float)kx - cx) * 0.125f) * row[kx];
            }
        }
        u[CR + c] = acc / (swy * swx);
    }
    __syncthreads();

    // patch-embed conv + relu
    {
        float acc = bb[t];
        const float* w = bw + (size_t)t * 192;
#pragma unroll 8
        for (int j = 0; j < 192; ++j) acc += patch[j] * w[j];
        u[t] = fmaxf(acc, 0.f);
    }
    if (t == 0) {
        float density = u[CR + 3], opacity = u[CR + 2], anis = u[CR + 4];
        float base = 0.5f * density + 0.25f * (1.f - opacity) + 0.25f * anis;
        wgt[p] = 1.f + 1.f / (1.f + expf(-4.f * (base - 0.5f)));
    }
    __syncthreads();

    // fusion GEMV row
    float fx = fb[t];
    const float* w2 = fw + (size_t)t * CF;
#pragma unroll 8
    for (int c = 0; c < CF; ++c) fx += w2[c] * u[c];
    xq[(size_t)p * KD + t] = fx;
    ahi[(size_t)p * KD + t] = bf16_rne(fx);
}

// ---------------------------------------------------------------------------
// Pass 1: approximate argmin_n (|m_n|^2 - 2 x.m_n) with bf16 WMMA.
// Grid: 64 M-blocks x 8 N-chunks. Each wave: 2 pinned A tiles (32 rows),
// one B fragment feeds two independent accumulator chains.
// ---------------------------------------------------------------------------
__global__ __launch_bounds__(256)
void nn_pass1_kernel(const unsigned short* __restrict__ ahi_g,
                     const unsigned short* __restrict__ bhi_g,
                     const float* __restrict__ bnorm,
                     unsigned long long* __restrict__ cand) {
    __shared__ unsigned long long sred[256][16];
    int tid = threadIdx.x;
    int wave = tid >> 5, lane = tid & 31;
    int half = lane >> 4;                 // 0: lanes 0-15, 1: lanes 16-31
    int mblk = blockIdx.x & (NMBLK - 1);
    int nchunk = blockIdx.x >> 6;
    int m0 = mblk * MBLK;

    const __bf16* Ahi = (const __bf16*)ahi_g;
    const __bf16* Bhi = (const __bf16*)bhi_g;

    // A fragments for two M-tiles (rows m0..m0+15, m0+16..m0+31), pinned.
    // 16x32 bf16 A, this code's contiguous k order (B loads match): lanes 0-15 K={k0..k0+15},
    //                    lanes 16-31 K={k0+16..k0+31}.
    v16bf a0[8], a1[8];
    {
        int ka = half ? 16 : 0;
        const __bf16* p0 = Ahi + (size_t)(m0 + (lane & 15)) * KD + ka;
        const __bf16* p1 = p0 + (size_t)16 * KD;
#pragma unroll
        for (int s = 0; s < 8; ++s) {
            union { v16bf v; v8bf h[2]; } t0, t1;
            t0.h[0] = *(const v8bf*)(p0 + 32 * s);
            t0.h[1] = *(const v8bf*)(p0 + 32 * s + 8);
            t1.h[0] = *(const v8bf*)(p1 + 32 * s);
            t1.h[1] = *(const v8bf*)(p1 + 32 * s + 8);
            a0[s] = t0.v;
            a1[s] = t1.v;
        }
    }

    float minv0[8], minv1[8];
    int arg0[8], arg1[8];
#pragma unroll
    for (int i = 0; i < 8; ++i) {
        minv0[i] = 3.402823466e38f; minv1[i] = 3.402823466e38f;
        arg0[i] = 0; arg1[i] = 0;
    }

    int kb = half ? 16 : 0;
    int t0base = nchunk * TILES_PER_CHUNK + wave * TILES_PER_WAVE;
    for (int it = 0; it < TILES_PER_WAVE; ++it) {
        int n0 = (t0base + it) << 4;
        int nrow = n0 + (lane & 15);
        const __bf16* pb = Bhi + (size_t)nrow * KD + kb;
        __builtin_prefetch(pb + (size_t)16 * KD, 0, 3);  // next tile's row
        v8f acc0 = {0.f, 0.f, 0.f, 0.f, 0.f, 0.f, 0.f, 0.f};
        v8f acc1 = {0.f, 0.f, 0.f, 0.f, 0.f, 0.f, 0.f, 0.f};
#pragma unroll
        for (int s = 0; s < 8; ++s) {
            v16bf bh = *(const v16bf*)(pb + 32 * s);
            acc0 = __builtin_amdgcn_wmma_f32_16x16x32_bf16(false, a0[s], false, bh, (short)0, acc0, false, false);
            acc1 = __builtin_amdgcn_wmma_f32_16x16x32_bf16(false, a1[s], false, bh, (short)0, acc1, false, false);
        }
        float bnv = bnorm[nrow];
#pragma unroll
        for (int i = 0; i < 8; ++i) {
            float k0 = bnv - 2.0f * acc0[i];
            float k1 = bnv - 2.0f * acc1[i];
            if (k0 < minv0[i]) { minv0[i] = k0; arg0[i] = nrow; }
            if (k1 < minv1[i]) { minv1[i] = k1; arg1[i] = nrow; }
        }
    }

    // pack (key, index); key = bnorm - 2*dot may be negative -> bias by
    // flipping sign bit so the u64 compare stays monotone.
#pragma unroll
    for (int i = 0; i < 8; ++i) {
        unsigned int b0 = __float_as_uint(minv0[i]);
        unsigned int b1 = __float_as_uint(minv1[i]);
        b0 = (b0 & 0x80000000u) ? ~b0 : (b0 | 0x80000000u);
        b1 = (b1 & 0x80000000u) ? ~b1 : (b1 | 0x80000000u);
        sred[tid][i]     = ((unsigned long long)b0 << 32) | (unsigned int)arg0[i];
        sred[tid][8 + i] = ((unsigned long long)b1 << 32) | (unsigned int)arg1[i];
    }
    __syncthreads();
    if (tid < MBLK) {
        // row r: chain c=r>>4, within-tile row rr=r&15 -> reg i=rr&7,
        // lane half = rr>>3.
        int r = tid;
        int c = r >> 4;
        int rr = r & 15;
        int i = (rr & 7) + c * 8;
        int base = (rr >> 3) ? 16 : 0;
        unsigned long long best = 0xFFFFFFFFFFFFFFFFull;
        for (int w = 0; w < 8; ++w)
            for (int l = 0; l < 16; ++l) {
                unsigned long long v = sred[w * 32 + base + l][i];
                best = (v < best) ? v : best;
            }
        atomicMin(&cand[m0 + r], best);
    }
}

// ---------------------------------------------------------------------------
// Pass 2: exact f32 distance to the winning bank row + outputs.
// ---------------------------------------------------------------------------
__global__ __launch_bounds__(256)
void nn_pass2_kernel(const float* __restrict__ xq,
                     const float* __restrict__ bank,
                     const unsigned long long* __restrict__ cand,
                     const float* __restrict__ wgt,
                     float* __restrict__ out) {
    __shared__ float red[256];
    int p = blockIdx.x, t = threadIdx.x;
    unsigned int n = (unsigned int)(cand[p] & 0xFFFFFFFFull);
    float d = xq[(size_t)p * KD + t] - bank[(size_t)n * KD + t];
    red[t] = d * d;
    __syncthreads();
    for (int s = 128; s > 0; s >>= 1) {
        if (t < s) red[t] += red[t + s];
        __syncthreads();
    }
    if (t == 0) {
        float s = sqrtf(fmaxf(red[0], 0.f));
        out[p] = s;                 // score_plain
        out[NPIX + p] = s * wgt[p]; // score_geo
    }
}

// ---------------------------------------------------------------------------
extern "C" void kernel_launch(void* const* d_in, const int* in_sizes, int n_in,
                              void* d_out, int out_size, void* d_ws, size_t ws_size,
                              hipStream_t stream) {
    const float* image = (const float*)d_in[0];
    const float* geo   = (const float*)d_in[1];
    const float* bw    = (const float*)d_in[2];
    const float* bb    = (const float*)d_in[3];
    const float* fw    = (const float*)d_in[4];
    const float* fb    = (const float*)d_in[5];
    const float* bank  = (const float*)d_in[6];
    float* out = (float*)d_out;
    (void)in_sizes; (void)n_in; (void)out_size; (void)ws_size;

    char* ws = (char*)d_ws;
    size_t off = 0;
    auto alloc = [&](size_t bytes) {
        char* ptr = ws + off;
        off = (off + bytes + 255) & ~(size_t)255;
        return ptr;
    };
    unsigned short* ahi      = (unsigned short*)alloc((size_t)NPIX * KD * 2);
    float* xq                = (float*)alloc((size_t)NPIX * KD * 4);
    float* wgt               = (float*)alloc((size_t)NPIX * 4);
    float* bnorm             = (float*)alloc((size_t)NBANK * 4);
    unsigned long long* cand = (unsigned long long*)alloc((size_t)NPIX * 8);
    unsigned short* bhi      = (unsigned short*)alloc((size_t)NBANK * KD * 2);

    init_kernel<<<(NPIX + 255) / 256, 256, 0, stream>>>(cand);
    bank_prep_kernel<<<NBANK, 256, 0, stream>>>(bank, bhi, bnorm);
    feature_kernel<<<NPIX, 256, 0, stream>>>(image, geo, bw, bb, fw, fb,
                                             xq, ahi, wgt);
    nn_pass1_kernel<<<NMBLK * NCHUNKS, 256, 0, stream>>>(ahi, bhi, bnorm, cand);
    nn_pass2_kernel<<<NPIX, 256, 0, stream>>>(xq, bank, cand, wgt, out);
}
